// MambaBlock_33389075759630
// MI455X (gfx1250) — matrix operand order repaired and checked by reading, not compiled, not run on hardware
//
#include <hip/hip_runtime.h>
#include <cmath>

#define DEV __device__ __forceinline__

typedef __attribute__((ext_vector_type(8)))  _Float16 half8;
typedef __attribute__((ext_vector_type(16))) _Float16 v16h;
typedef __attribute__((ext_vector_type(8)))  float    v8f;

// Problem dimensions (from reference setup_inputs)
static constexpr int Bb   = 8;
static constexpr int L    = 2048;
static constexpr int Di   = 256;     // d_inner
static constexpr int Ns   = 16;      // state size n
static constexpr int DtR  = 8;       // dt_rank
static constexpr int Kk   = 8;       // spline coeffs per input (GRID+ORDER)
static constexpr int Ntok = Bb * L;  // 16384
static constexpr int KC   = 288;     // feature chunk = 32 inputs * 9 features
static constexpr int MR   = 64;      // rows (tokens) per block
static constexpr int CH   = 128;     // scan chunk length
static constexpr int NCH  = L / CH;  // 16 chunks

DEV float siluf(float x) { return x / (1.0f + expf(-x)); }

// 8 cubic B-spline bases on uniform grid: grid[j] = (j-3)*0.4 - 1, j=0..11
DEV void spline8(float x, float* out) {
  const float h = 0.4f;
  float g[12];
  #pragma unroll
  for (int j = 0; j < 12; ++j) g[j] = (float)(j - 3) * h - 1.0f;
  float bb[11];
  #pragma unroll
  for (int j = 0; j < 11; ++j) bb[j] = (x >= g[j] && x < g[j + 1]) ? 1.0f : 0.0f;
  #pragma unroll
  for (int k = 1; k <= 3; ++k) {
    float inv = 1.0f / ((float)k * h);
    #pragma unroll
    for (int j = 0; j + k < 11; ++j)
      bb[j] = (x - g[j]) * inv * bb[j] + (g[j + k + 1] - x) * inv * bb[j + 1];
  }
  #pragma unroll
  for (int j = 0; j < 8; ++j) out[j] = bb[j];
}

DEV void hilo(float f, _Float16& hi, _Float16& lo) {
  hi = (_Float16)f;
  lo = (_Float16)(f - (float)hi);
}

// Fold KAN weights: W[o][i*9+0]=base_w[o][i]; W[o][i*9+1+k]=spline_w[o][i][k]*scaler[o][i]
// mode==1 (plain): W[o][col]=base_w[o][col]. Pads (rows>=d_out_real, cols>=valid) are zero.
__global__ void fold_weights(const float* __restrict__ basew,
                             const float* __restrict__ splinew,
                             const float* __restrict__ scaler,
                             _Float16* __restrict__ Wh, _Float16* __restrict__ Wl,
                             int d_out_pad, int d_out_real, int d_in, int Kp, int mode)
{
  int idx = blockIdx.x * blockDim.x + threadIdx.x;
  int total = d_out_pad * Kp;
  if (idx >= total) return;
  int o = idx / Kp, col = idx % Kp;
  float v = 0.0f;
  if (o < d_out_real) {
    if (mode == 0) {
      if (col < 9 * d_in) {
        int i = col / 9, j = col % 9;
        v = (j == 0) ? basew[o * d_in + i]
                     : splinew[(o * d_in + i) * Kk + (j - 1)] * scaler[o * d_in + i];
      }
    } else {
      if (col < d_in) v = basew[o * d_in + col];
    }
  }
  _Float16 hh, ll; hilo(v, hh, ll);
  Wh[idx] = hh; Wl[idx] = ll;
}

// Fused KAN-featurize + hi/lo f16 WMMA GEMM.
// Block = 512 threads = 16 waves = 4 row-subtiles x 4 col-groups; 64 tokens/block.
// Wave (rowgrp, colgrp): row tile rowgrp, col tiles {colgrp, colgrp+4, ...} (NT of them).
// MODE 0: KAN featurization (silu + 8 splines per input, interleaved i*9+{0..8});
// MODE 1: plain copy. epilogue 1: softplus.
template<int NT, int KP, int DPAD, int MODE>
__global__ __launch_bounds__(512)
void kan_gemm(const float* __restrict__ X, int ldx, int d_in,
              const _Float16* __restrict__ Wh, const _Float16* __restrict__ Wl,
              int d_out_real, float* __restrict__ Y, int ld_out, int epilogue)
{
  __shared__ __align__(32) _Float16 sH[MR * KC];
  __shared__ __align__(32) _Float16 sL[MR * KC];
  const int tid    = threadIdx.x;
  const int wave_s = __builtin_amdgcn_readfirstlane(tid) >> 5;  // scalar wave id
  const int rowgrp = wave_s >> 2;   // 0..3 (scalar)
  const int colgrp = wave_s & 3;    // 0..3 (scalar)
  const int lane   = tid & 31;
  const int m      = lane & 15;     // A row / B col / D col
  const int hh     = lane >> 4;     // lane half
  const long rowblk = (long)blockIdx.x * MR;

  v8f acc[NT];
  #pragma unroll
  for (int ci = 0; ci < NT; ++ci) { v8f z = {}; acc[ci] = z; }

  for (int kc = 0; kc < KP; kc += KC) {
    __syncthreads();
    if (MODE == 0) {
      const int i0 = kc / 9;  // chunk covers inputs [i0, i0+32)
      for (int task = tid; task < MR * 32; task += 512) {
        int rr = task & (MR - 1);
        int ii = i0 + (task >> 6);
        float v[9];
        if (ii < d_in) {
          float xv = X[(rowblk + rr) * ldx + ii];
          v[0] = siluf(xv);
          spline8(xv, v + 1);
        } else {
          #pragma unroll
          for (int j = 0; j < 9; ++j) v[j] = 0.0f;
        }
        int cb = ii * 9 - kc;
        #pragma unroll
        for (int j = 0; j < 9; ++j) {
          _Float16 hv, lv; hilo(v[j], hv, lv);
          sH[rr * KC + cb + j] = hv;
          sL[rr * KC + cb + j] = lv;
        }
      }
    } else {
      for (int task = tid; task < MR * KC; task += 512) {
        int rr = task & (MR - 1);
        int cl = task >> 6;
        int col = kc + cl;
        float xv = (col < d_in) ? X[(rowblk + rr) * ldx + col] : 0.0f;
        _Float16 hv, lv; hilo(xv, hv, lv);
        sH[rr * KC + cl] = hv;
        sL[rr * KC + cl] = lv;
      }
    }
    __syncthreads();

    const int kend = (KP - kc < KC) ? (KP - kc) : KC;
    for (int ks = 0; ks < kend; ks += 32) {
      // A 16x32 f16 as loaded here: lane half h holds K = 16h..16h+15 contiguous (same order as B)
      const _Float16* pH = &sH[(rowgrp * 16 + m) * KC + ks + 16 * hh];
      const _Float16* pL = &sL[(rowgrp * 16 + m) * KC + ks + 16 * hh];
      half8 aH0 = *(const half8*)pH;
      half8 aH1 = *(const half8*)(pH + 8);
      half8 aL0 = *(const half8*)pL;
      half8 aL1 = *(const half8*)(pL + 8);
      v16h ah = __builtin_shufflevector(aH0, aH1, 0,1,2,3,4,5,6,7,8,9,10,11,12,13,14,15);
      v16h al = __builtin_shufflevector(aL0, aL1, 0,1,2,3,4,5,6,7,8,9,10,11,12,13,14,15);
      #pragma unroll
      for (int ci = 0; ci < NT; ++ci) {
        const int ct = colgrp + 4 * ci;
        if (ct * 16 < DPAD) {   // scalar branch (colgrp is SALU, DPAD constexpr)
          // B 32x16 f16 as loaded here: lane = col n, half h holds K = 16h..16h+15 contiguous (same as A)
          const size_t off = (size_t)(ct * 16 + m) * KP + (size_t)(kc + ks) + 16 * hh;
          v16h bh = *(const v16h*)(Wh + off);
          v16h bl = *(const v16h*)(Wl + off);
          acc[ci] = __builtin_amdgcn_wmma_f32_16x16x32_f16(false, ah, false, bh, (short)0, acc[ci], false, false);
          acc[ci] = __builtin_amdgcn_wmma_f32_16x16x32_f16(false, ah, false, bl, (short)0, acc[ci], false, false);
          acc[ci] = __builtin_amdgcn_wmma_f32_16x16x32_f16(false, al, false, bh, (short)0, acc[ci], false, false);
        }
      }
    }
  }

  // D layout: VGPR r -> row (r + 8*half), col = lane&15
  #pragma unroll
  for (int ci = 0; ci < NT; ++ci) {
    const int ct = colgrp + 4 * ci;
    if (ct * 16 >= DPAD) continue;
    int col = ct * 16 + m;
    if (col < d_out_real) {
      #pragma unroll
      for (int r = 0; r < 8; ++r) {
        long row = rowblk + rowgrp * 16 + r + 8 * hh;
        float v = acc[ci][r];
        if (epilogue == 1) v = (v > 20.0f) ? v : log1pf(expf(v));
        Y[row * (long)ld_out + col] = v;
      }
    }
  }
}

// Depthwise causal conv1d (k=4, left pad 3) + SiLU. Reads xs = xr[:, :256] (stride 512).
__global__ void conv_silu_kernel(const float* __restrict__ xr,
                                 const float* __restrict__ cw,
                                 const float* __restrict__ cb,
                                 float* __restrict__ xc)
{
  long idx = (long)blockIdx.x * blockDim.x + threadIdx.x;
  if (idx >= (long)Ntok * Di) return;
  int c   = (int)(idx % Di);
  long tok = idx / Di;
  int t   = (int)(tok % L);
  long b  = tok / L;
  float s = cb[c];
  #pragma unroll
  for (int j = 0; j < 4; ++j) {
    int tt = t - 3 + j;
    if (tt >= 0) s += cw[c * 4 + j] * xr[(b * L + tt) * 512 + c];
  }
  xc[tok * Di + c] = siluf(s);
}

// Selective scan phase 1: per (b, chunk, d) compute P[n]=prod(dA), S[n]=chunk-local state.
__global__ __launch_bounds__(256)
void scan_phase1(const float* __restrict__ delta, const float* __restrict__ xc,
                 const float* __restrict__ xdbl, const float* __restrict__ A_log,
                 float* __restrict__ chP, float* __restrict__ chS)
{
  int b = blockIdx.x / NCH;
  int c = blockIdx.x % NCH;
  int d = threadIdx.x;
  float Ad[Ns], P[Ns], S[Ns];
  #pragma unroll
  for (int n = 0; n < Ns; ++n) { Ad[n] = -expf(A_log[d * Ns + n]); P[n] = 1.0f; S[n] = 0.0f; }
  long t0 = (long)b * L + (long)c * CH;
  for (int t = 0; t < CH; ++t) {
    long row = t0 + t;
    float dt = delta[row * Di + d];
    float u  = xc[row * Di + d];
    float du = dt * u;
    const float* Brow = xdbl + row * 40 + DtR;
    #pragma unroll
    for (int n = 0; n < Ns; ++n) {
      float a = expf(dt * Ad[n]);
      P[n] *= a;
      S[n] = a * S[n] + du * Brow[n];
    }
  }
  long base = (((long)b * NCH + c) * Di + d) * Ns;
  #pragma unroll
  for (int n = 0; n < Ns; ++n) { chP[base + n] = P[n]; chS[base + n] = S[n]; }
}

// Phase 2: combine chunks sequentially per (b,d,n): emit incoming state for each chunk.
__global__ void scan_phase2(const float* __restrict__ chP, const float* __restrict__ chS,
                            float* __restrict__ inState)
{
  int idx = blockIdx.x * blockDim.x + threadIdx.x;
  if (idx >= Bb * Di * Ns) return;
  int n = idx & 15;
  int d = (idx >> 4) & (Di - 1);
  int b = idx >> 12;
  float s = 0.0f;
  for (int c = 0; c < NCH; ++c) {
    long o = (((long)b * NCH + c) * Di + d) * Ns + n;
    inState[o] = s;
    s = chP[o] * s + chS[o];
  }
}

// Phase 3: replay chunk from incoming state, emit z = (y + u*D) * silu(res).
__global__ __launch_bounds__(256)
void scan_phase3(const float* __restrict__ delta, const float* __restrict__ xc,
                 const float* __restrict__ xdbl, const float* __restrict__ A_log,
                 const float* __restrict__ Dvec, const float* __restrict__ xr,
                 const float* __restrict__ inState, float* __restrict__ z)
{
  int b = blockIdx.x / NCH;
  int c = blockIdx.x % NCH;
  int d = threadIdx.x;
  float Ad[Ns], s[Ns];
  long base = (((long)b * NCH + c) * Di + d) * Ns;
  #pragma unroll
  for (int n = 0; n < Ns; ++n) { Ad[n] = -expf(A_log[d * Ns + n]); s[n] = inState[base + n]; }
  float Dd = Dvec[d];
  long t0 = (long)b * L + (long)c * CH;
  for (int t = 0; t < CH; ++t) {
    long row = t0 + t;
    float dt = delta[row * Di + d];
    float u  = xc[row * Di + d];
    float du = dt * u;
    const float* Brow = xdbl + row * 40 + DtR;
    const float* Crow = xdbl + row * 40 + DtR + Ns;
    float y = 0.0f;
    #pragma unroll
    for (int n = 0; n < Ns; ++n) {
      float a = expf(dt * Ad[n]);
      s[n] = a * s[n] + du * Brow[n];
      y += s[n] * Crow[n];
    }
    float res = xr[row * 512 + Di + d];
    z[row * Di + d] = (y + u * Dd) * siluf(res);
  }
}

extern "C" void kernel_launch(void* const* d_in, const int* in_sizes, int n_in,
                              void* d_out, int out_size, void* d_ws, size_t ws_size,
                              hipStream_t stream)
{
  (void)in_sizes; (void)n_in; (void)out_size; (void)ws_size;
  const float* x      = (const float*)d_in[0];
  const float* in_bw  = (const float*)d_in[1];
  const float* in_sw  = (const float*)d_in[2];
  const float* in_sc  = (const float*)d_in[3];
  const float* conv_w = (const float*)d_in[4];
  const float* conv_b = (const float*)d_in[5];
  const float* xp_bw  = (const float*)d_in[6];
  const float* xp_sw  = (const float*)d_in[7];
  const float* xp_sc  = (const float*)d_in[8];
  const float* dt_bw  = (const float*)d_in[9];
  const float* dt_sw  = (const float*)d_in[10];
  const float* dt_sc  = (const float*)d_in[11];
  const float* A_log  = (const float*)d_in[12];
  const float* Dvec   = (const float*)d_in[13];
  const float* out_w  = (const float*)d_in[14];

  char* p = (char*)d_ws;
  auto alloc = [&](size_t bytes) -> char* {
    char* r = p; p += (bytes + 255) & ~(size_t)255; return r;
  };
  _Float16* W1h = (_Float16*)alloc((size_t)512 * 1152 * 2);
  _Float16* W1l = (_Float16*)alloc((size_t)512 * 1152 * 2);
  _Float16* W2h = (_Float16*)alloc((size_t)48 * 2304 * 2);
  _Float16* W2l = (_Float16*)alloc((size_t)48 * 2304 * 2);
  _Float16* W3h = (_Float16*)alloc((size_t)256 * 96 * 2);
  _Float16* W3l = (_Float16*)alloc((size_t)256 * 96 * 2);
  _Float16* WOh = (_Float16*)alloc((size_t)128 * 256 * 2);
  _Float16* WOl = (_Float16*)alloc((size_t)128 * 256 * 2);
  float* xr    = (float*)alloc((size_t)Ntok * 512 * 4);
  float* xc    = (float*)alloc((size_t)Ntok * Di * 4);
  float* xdbl  = (float*)alloc((size_t)Ntok * 40 * 4);
  float* dlt   = (float*)alloc((size_t)Ntok * Di * 4);
  float* chP   = (float*)alloc((size_t)Bb * NCH * Di * Ns * 4);
  float* chS   = (float*)alloc((size_t)Bb * NCH * Di * Ns * 4);
  float* inSt  = (float*)alloc((size_t)Bb * NCH * Di * Ns * 4);
  float* zbuf  = (float*)alloc((size_t)Ntok * Di * 4);

  const int T = 256;
  // Fold weights (with scaler) into interleaved hi/lo f16 matrices
  fold_weights<<<(512 * 1152 + T - 1) / T, T, 0, stream>>>(in_bw, in_sw, in_sc, W1h, W1l, 512, 512, 128, 1152, 0);
  fold_weights<<<(48 * 2304 + T - 1) / T, T, 0, stream>>>(xp_bw, xp_sw, xp_sc, W2h, W2l, 48, 40, 256, 2304, 0);
  fold_weights<<<(256 * 96 + T - 1) / T, T, 0, stream>>>(dt_bw, dt_sw, dt_sc, W3h, W3l, 256, 256, 8, 96, 0);
  fold_weights<<<(128 * 256 + T - 1) / T, T, 0, stream>>>(out_w, nullptr, nullptr, WOh, WOl, 128, 128, 256, 256, 1);

  const int GB = Ntok / MR;  // 256 blocks
  // kan_linear 1: x (N x 128) -> x_and_res (N x 512)
  kan_gemm<8, 1152, 512, 0><<<GB, 512, 0, stream>>>(x, 128, 128, W1h, W1l, 512, xr, 512, 0);
  // depthwise causal conv + silu on xs half
  conv_silu_kernel<<<((long)Ntok * Di + T - 1) / T, T, 0, stream>>>(xr, conv_w, conv_b, xc);
  // kan_linear 2: xc (N x 256) -> x_dbl (N x 40)
  kan_gemm<1, 2304, 48, 0><<<GB, 512, 0, stream>>>(xc, Di, Di, W2h, W2l, 40, xdbl, 40, 0);
  // kan_linear 3 + softplus: delta_in (N x 8, stride 40) -> delta (N x 256)
  kan_gemm<4, 96, 256, 0><<<GB, 512, 0, stream>>>(xdbl, 40, 8, W3h, W3l, 256, dlt, 256, 1);
  // chunked selective scan
  scan_phase1<<<Bb * NCH, 256, 0, stream>>>(dlt, xc, xdbl, A_log, chP, chS);
  scan_phase2<<<(Bb * Di * Ns + T - 1) / T, T, 0, stream>>>(chP, chS, inSt);
  scan_phase3<<<Bb * NCH, 256, 0, stream>>>(dlt, xc, xdbl, A_log, Dvec, xr, inSt, zbuf);
  // final projection: z (N x 256) @ out_w^T -> out (N x 128)
  kan_gemm<2, 256, 128, 1><<<GB, 512, 0, stream>>>(zbuf, Di, Di, WOh, WOl, 128, (float*)d_out, 128, 0);
}
